// QwenMambaLayer_88390426952210
// MI455X (gfx1250) — compile-verified, matrix operand order checked by reading, not run on hardware
//
#include <hip/hip_runtime.h>
#include <cmath>

// ---------------- types ----------------
typedef __bf16 bf16_t;
typedef __attribute__((ext_vector_type(16))) __bf16 v16bf;
typedef __attribute__((ext_vector_type(8)))  __bf16 v8bf;
typedef __attribute__((ext_vector_type(8)))  float  v8f;

// ---------------- problem dims ----------------
#define D_MODEL 1024
#define D_INNER 2048
#define D_STATE 16
#define D_CONV  4
#define DT_RANK 64
#define BATCH   2
#define SEQ     2048
#define TOKENS  (BATCH * SEQ)           // 4096
#define XP_COLS (DT_RANK + 2 * D_STATE) // 96

// =====================================================================
// Generic bf16 WMMA GEMM:  C[M,N] = A[M,K](row) * BT[N,K](row-major B^T)
// Optionally C += addC (fused residual). Wave tile 32x32 (2x2 WMMA),
// block = 8 waves -> 64(M) x 128(N) tile. K multiple of 64 not required;
// K/32 assumed even here (true for all launches: 32, 64, 2 tiles).
// OOB N tiles: loads are index-clamped (branch-free), stores guarded.
// =====================================================================
__global__ __launch_bounds__(256)
void gemm_bf16_wmma(const bf16_t* __restrict__ A, int lda,
                    const bf16_t* __restrict__ BT, int ldb,
                    float* __restrict__ C, int ldc,
                    const float* __restrict__ addC,
                    int M, int N, int K)
{
    const int lane  = threadIdx.x & 31;
    const int wave  = threadIdx.x >> 5;
    const int waveM = wave >> 2;            // 0..1
    const int waveN = wave & 3;             // 0..3
    const int mBase = blockIdx.y * 64  + waveM * 32;
    const int nBase = blockIdx.x * 128 + waveN * 32;
    const int half  = lane >> 4;            // 0/1
    const int l16   = lane & 15;

    v8f acc[2][2];
    #pragma unroll
    for (int i = 0; i < 2; i++)
        #pragma unroll
        for (int j = 0; j < 2; j++)
            #pragma unroll
            for (int r = 0; r < 8; r++) acc[i][j][r] = 0.0f;

    const bool nOk0 = (nBase)      < N;     // N multiple of 16
    const bool nOk1 = (nBase + 16) < N;

    // Branch-free OOB handling: clamp B rows into range; OOB results unused.
    const int nRow0 = min(nBase +      l16, N - 1);
    const int nRow1 = min(nBase + 16 + l16, N - 1);

    // Hoisted fragment base pointers (ISA 16-bit A 16x32 lane layout:
    // elements 0..7 at k = half*8 + e, elements 8..15 at k = 16 + half*8 + e).
    const bf16_t* pa0 = A  + (size_t)(mBase +      l16) * lda + half * 8;
    const bf16_t* pa1 = A  + (size_t)(mBase + 16 + l16) * lda + half * 8;
    const bf16_t* pb0 = BT + (size_t)nRow0 * ldb + half * 8;
    const bf16_t* pb1 = BT + (size_t)nRow1 * ldb + half * 8;

    const int kTiles = K >> 5;
    #pragma unroll 2
    for (int kt = 0; kt < kTiles; ++kt) {
        const int ko = kt * 32;
        v16bf a[2], b[2];
        {
            v8bf lo = *(const v8bf*)(pa0 + ko);
            v8bf hi = *(const v8bf*)(pa0 + ko + 16);
            #pragma unroll
            for (int e = 0; e < 8; e++) { a[0][e] = lo[e]; a[0][8 + e] = hi[e]; }
        }
        {
            v8bf lo = *(const v8bf*)(pa1 + ko);
            v8bf hi = *(const v8bf*)(pa1 + ko + 16);
            #pragma unroll
            for (int e = 0; e < 8; e++) { a[1][e] = lo[e]; a[1][8 + e] = hi[e]; }
        }
        {
            v8bf lo = *(const v8bf*)(pb0 + ko);
            v8bf hi = *(const v8bf*)(pb0 + ko + 16);
            #pragma unroll
            for (int e = 0; e < 8; e++) { b[0][e] = lo[e]; b[0][8 + e] = hi[e]; }
        }
        {
            v8bf lo = *(const v8bf*)(pb1 + ko);
            v8bf hi = *(const v8bf*)(pb1 + ko + 16);
            #pragma unroll
            for (int e = 0; e < 8; e++) { b[1][e] = lo[e]; b[1][8 + e] = hi[e]; }
        }
        if (kt + 1 < kTiles) {              // global_prefetch_b8 next K tile
            __builtin_prefetch(pa0 + ko + 32, 0, 0);
            __builtin_prefetch(pb0 + ko + 32, 0, 0);
        }
        #pragma unroll
        for (int i = 0; i < 2; i++)
            #pragma unroll
            for (int j = 0; j < 2; j++)
                acc[i][j] = __builtin_amdgcn_wmma_f32_16x16x32_bf16(
                    false, a[i], false, b[j], (short)0, acc[i][j], false, false);
    }

    // D layout: VGPR r -> M = r (lanes 0-15) / 8+r (lanes 16-31), N = lane%16
    #pragma unroll
    for (int i = 0; i < 2; i++)
        #pragma unroll
        for (int j = 0; j < 2; j++) {
            const bool ok = j ? nOk1 : nOk0;
            if (!ok) continue;
            const int n = nBase + j * 16 + l16;
            #pragma unroll
            for (int r = 0; r < 8; r++) {
                const int m = mBase + i * 16 + half * 8 + r;
                float v = acc[i][j][r];
                if (addC) v += addC[(size_t)m * ldc + n];
                C[(size_t)m * ldc + n] = v;
            }
        }
}

// =====================================================================
// Weight transpose + f32 -> bf16:  WT[n*K + k] = (bf16) W[k*N + n]
// =====================================================================
__global__ void transpose_to_bf16(const float* __restrict__ W,
                                  bf16_t* __restrict__ WT, int K, int N)
{
    size_t idx = (size_t)blockIdx.x * 256 + threadIdx.x;
    if (idx >= (size_t)K * N) return;
    int n = (int)(idx / K);
    int k = (int)(idx % K);
    WT[idx] = (bf16_t)W[(size_t)k * N + n];
}

// =====================================================================
// RMSNorm row kernel: one block per token row, output bf16
// =====================================================================
__global__ __launch_bounds__(256)
void rmsnorm_kernel(const float* __restrict__ x, const float* __restrict__ w,
                    bf16_t* __restrict__ xn)
{
    __shared__ float red[256];
    const int row = blockIdx.x;                  // 0..TOKENS-1
    const float* xr = x + (size_t)row * D_MODEL;
    float s = 0.f;
    for (int c = threadIdx.x; c < D_MODEL; c += 256) { float v = xr[c]; s += v * v; }
    red[threadIdx.x] = s;
    __syncthreads();
    for (int off = 128; off > 0; off >>= 1) {
        if (threadIdx.x < off) red[threadIdx.x] += red[threadIdx.x + off];
        __syncthreads();
    }
    const float inv = rsqrtf(red[0] * (1.0f / D_MODEL) + 1e-6f);
    for (int c = threadIdx.x; c < D_MODEL; c += 256)
        xn[(size_t)row * D_MODEL + c] = (bf16_t)(xr[c] * inv * w[c]);
}

// =====================================================================
// Causal depthwise conv (taps=4) + SiLU; reads x-half of xz, writes bf16
// =====================================================================
__global__ void conv_silu_kernel(const float* __restrict__ xz,
                                 const float* __restrict__ cw,
                                 const float* __restrict__ cb,
                                 bf16_t* __restrict__ xc_bf)
{
    size_t idx = (size_t)blockIdx.x * 256 + threadIdx.x;
    if (idx >= (size_t)TOKENS * D_INNER) return;
    int d = (int)(idx % D_INNER);
    int t = (int)((idx / D_INNER) % SEQ);
    int b = (int)(idx / ((size_t)D_INNER * SEQ));
    float acc = cb[d];
    #pragma unroll
    for (int j = 0; j < D_CONV; j++) {
        int tt = t - (D_CONV - 1) + j;
        if (tt >= 0)
            acc += cw[d * D_CONV + j] *
                   xz[((size_t)b * SEQ + tt) * (2 * D_INNER) + d];
    }
    float sl = acc / (1.f + __expf(-acc));       // silu
    xc_bf[idx] = (bf16_t)sl;
}

// =====================================================================
// Slice dt columns (stride 96) out of xp into contiguous bf16 [TOKENS,64]
// =====================================================================
__global__ void dtslice_kernel(const float* __restrict__ xp,
                               bf16_t* __restrict__ dt_bf)
{
    size_t idx = (size_t)blockIdx.x * 256 + threadIdx.x;
    if (idx >= (size_t)TOKENS * DT_RANK) return;
    int c = (int)(idx % DT_RANK);
    size_t r = idx / DT_RANK;
    dt_bf[idx] = (bf16_t)xp[r * XP_COLS + c];
}

// =====================================================================
// delta = softplus(delta_raw + dt_b), in place
// =====================================================================
__global__ void softplus_kernel(float* __restrict__ delta,
                                const float* __restrict__ dt_b)
{
    size_t idx = (size_t)blockIdx.x * 256 + threadIdx.x;
    if (idx >= (size_t)TOKENS * D_INNER) return;
    int d = (int)(idx % D_INNER);
    float v = delta[idx] + dt_b[d];
    delta[idx] = (v > 20.f) ? v : log1pf(__expf(v));
}

// =====================================================================
// Selective scan: one thread per (b, channel). 16-wide state in regs.
// Fuses  y = (scan + xc*D) * silu(z)  and writes bf16 for out_proj.
// =====================================================================
__global__ __launch_bounds__(256)
void scan_kernel(const float* __restrict__ delta,
                 const float* __restrict__ xp,
                 const bf16_t* __restrict__ xc_bf,
                 const float* __restrict__ xz,
                 const float* __restrict__ logA,
                 const float* __restrict__ Dp,
                 bf16_t* __restrict__ y_bf)
{
    int tid = blockIdx.x * 256 + threadIdx.x;    // 0..BATCH*D_INNER-1
    if (tid >= BATCH * D_INNER) return;
    int b = tid / D_INNER;
    int d = tid % D_INNER;
    float Arow[D_STATE], h[D_STATE];
    #pragma unroll
    for (int s = 0; s < D_STATE; s++) {
        Arow[s] = -__expf(logA[d * D_STATE + s]);
        h[s] = 0.f;
    }
    const float Dd = Dp[d];
    for (int t = 0; t < SEQ; t++) {
        size_t tok = (size_t)b * SEQ + t;
        float dl = delta[tok * D_INNER + d];
        float xc = (float)xc_bf[tok * D_INNER + d];
        const float* xpr = xp + tok * XP_COLS;
        float y = 0.f;
        #pragma unroll
        for (int s = 0; s < D_STATE; s++) {
            h[s] = __expf(dl * Arow[s]) * h[s] + dl * xpr[DT_RANK + s] * xc;
            y += h[s] * xpr[DT_RANK + D_STATE + s];
        }
        float z = xz[tok * (2 * D_INNER) + D_INNER + d];
        y = (y + xc * Dd) * (z / (1.f + __expf(-z)));
        y_bf[tok * D_INNER + d] = (bf16_t)y;
    }
}

// =====================================================================
// launcher
// =====================================================================
extern "C" void kernel_launch(void* const* d_in, const int* in_sizes, int n_in,
                              void* d_out, int out_size, void* d_ws, size_t ws_size,
                              hipStream_t stream)
{
    (void)in_sizes; (void)n_in; (void)out_size; (void)ws_size;

    const float* x       = (const float*)d_in[0];   // [B,L,1024]
    const float* norm_w  = (const float*)d_in[1];   // [1024]
    const float* in_w    = (const float*)d_in[2];   // [1024, 4096]
    const float* conv_w  = (const float*)d_in[3];   // [2048,1,4]
    const float* conv_b  = (const float*)d_in[4];   // [2048]
    const float* xproj_w = (const float*)d_in[5];   // [2048, 96]
    const float* dt_w    = (const float*)d_in[6];   // [64, 2048]
    const float* dt_b    = (const float*)d_in[7];   // [2048]
    const float* log_A   = (const float*)d_in[8];   // [2048,16]
    const float* D_param = (const float*)d_in[9];   // [2048]
    const float* out_w   = (const float*)d_in[10];  // [2048, 1024]
    float* out = (float*)d_out;                     // [B,L,1024] f32

    // ---- workspace carve-out (256B aligned) ----
    char* ws = (char*)d_ws;
    size_t off = 0;
    auto alloc = [&](size_t bytes) -> char* {
        char* p = ws + off;
        off = (off + bytes + 255) & ~(size_t)255;
        return p;
    };
    bf16_t* w1T   = (bf16_t*)alloc(sizeof(bf16_t) * (size_t)(2 * D_INNER) * D_MODEL); // 4096x1024
    bf16_t* w2T   = (bf16_t*)alloc(sizeof(bf16_t) * (size_t)XP_COLS * D_INNER);       // 96x2048
    bf16_t* w3T   = (bf16_t*)alloc(sizeof(bf16_t) * (size_t)D_INNER * DT_RANK);       // 2048x64
    bf16_t* w4T   = (bf16_t*)alloc(sizeof(bf16_t) * (size_t)D_MODEL * D_INNER);       // 1024x2048
    bf16_t* xn_bf = (bf16_t*)alloc(sizeof(bf16_t) * (size_t)TOKENS * D_MODEL);
    float*  xz    = (float*) alloc(sizeof(float)  * (size_t)TOKENS * 2 * D_INNER);
    bf16_t* xc_bf = (bf16_t*)alloc(sizeof(bf16_t) * (size_t)TOKENS * D_INNER);
    float*  xp    = (float*) alloc(sizeof(float)  * (size_t)TOKENS * XP_COLS);
    bf16_t* dt_bf = (bf16_t*)alloc(sizeof(bf16_t) * (size_t)TOKENS * DT_RANK);
    float*  delta = (float*) alloc(sizeof(float)  * (size_t)TOKENS * D_INNER);
    bf16_t* y_bf  = (bf16_t*)alloc(sizeof(bf16_t) * (size_t)TOKENS * D_INNER);

    auto grid1 = [](size_t n) { return dim3((unsigned)((n + 255) / 256)); };

    // 1) weight transposes -> bf16 W^T
    transpose_to_bf16<<<grid1((size_t)D_MODEL * 2 * D_INNER), 256, 0, stream>>>(in_w,   w1T, D_MODEL, 2 * D_INNER);
    transpose_to_bf16<<<grid1((size_t)D_INNER * XP_COLS),     256, 0, stream>>>(xproj_w, w2T, D_INNER, XP_COLS);
    transpose_to_bf16<<<grid1((size_t)DT_RANK * D_INNER),     256, 0, stream>>>(dt_w,   w3T, DT_RANK, D_INNER);
    transpose_to_bf16<<<grid1((size_t)D_INNER * D_MODEL),     256, 0, stream>>>(out_w,  w4T, D_INNER, D_MODEL);

    // 2) RMSNorm -> bf16
    rmsnorm_kernel<<<TOKENS, 256, 0, stream>>>(x, norm_w, xn_bf);

    // 3) in_proj: xz[4096,4096] = xn @ in_w
    gemm_bf16_wmma<<<dim3((2 * D_INNER) / 128, TOKENS / 64), 256, 0, stream>>>(
        xn_bf, D_MODEL, w1T, D_MODEL, xz, 2 * D_INNER, nullptr,
        TOKENS, 2 * D_INNER, D_MODEL);

    // 4) causal depthwise conv + silu -> bf16
    conv_silu_kernel<<<grid1((size_t)TOKENS * D_INNER), 256, 0, stream>>>(xz, conv_w, conv_b, xc_bf);

    // 5) x_proj: xp[4096,96] = xc @ xproj_w   (N=96 -> clamped/guarded tiles)
    gemm_bf16_wmma<<<dim3(1, TOKENS / 64), 256, 0, stream>>>(
        xc_bf, D_INNER, w2T, D_INNER, xp, XP_COLS, nullptr,
        TOKENS, XP_COLS, D_INNER);

    // 6) slice dt -> contiguous bf16
    dtslice_kernel<<<grid1((size_t)TOKENS * DT_RANK), 256, 0, stream>>>(xp, dt_bf);

    // 7) dt_proj: delta_raw[4096,2048] = dt @ dt_w
    gemm_bf16_wmma<<<dim3(D_INNER / 128, TOKENS / 64), 256, 0, stream>>>(
        dt_bf, DT_RANK, w3T, DT_RANK, delta, D_INNER, nullptr,
        TOKENS, D_INNER, DT_RANK);

    // 8) delta = softplus(delta_raw + dt_b)
    softplus_kernel<<<grid1((size_t)TOKENS * D_INNER), 256, 0, stream>>>(delta, dt_b);

    // 9) selective scan (+ D skip + silu gate), writes bf16 y
    scan_kernel<<<dim3((BATCH * D_INNER) / 256), 256, 0, stream>>>(
        delta, xp, xc_bf, xz, log_A, D_param, y_bf);

    // 10) out_proj with fused residual: out = x + y @ out_w
    gemm_bf16_wmma<<<dim3(D_MODEL / 128, TOKENS / 64), 256, 0, stream>>>(
        y_bf, D_INNER, w4T, D_INNER, out, D_MODEL, x,
        TOKENS, D_MODEL, D_INNER);
}
